// MHA_CrossAttention_RelBias_54417235640936
// MI455X (gfx1250) — matrix operand order repaired and checked by reading, compiled, not run on hardware
//
#include <hip/hip_runtime.h>

typedef __bf16 bhalf_t;
typedef __bf16 v16bf __attribute__((ext_vector_type(16)));
typedef __bf16 v8bf  __attribute__((ext_vector_type(8)));
typedef float  v8f   __attribute__((ext_vector_type(8)));

// Problem constants (from reference)
constexpr int Bn    = 4;
constexpr int CH    = 256;
constexpr int CTX   = 512;
constexpr int Hh    = 48;
constexpr int Ww    = 48;
constexpr int S     = Hh * Ww;          // 2304
constexpr int HEADS = 8;
constexpr int HD    = 64;
constexpr int INNER = HEADS * HD;       // 512
constexpr int QT    = S / 16;           // 144 query tiles

// ---------------- WMMA helpers ----------------

__device__ inline v8f wmma_bf16(v16bf a, v16bf b, v8f c) {
  return __builtin_amdgcn_wmma_f32_16x16x32_bf16(
      /*neg_a=*/false, a, /*neg_b=*/false, b,
      /*c_mod=*/(short)0, c, /*reuse_a=*/false, /*reuse_b=*/false);
}

// A-matrix 16x32 bf16 fragment. ISA layout:
//  lanes 0-15 : row M=lane,   elems 0-7 -> K=0..7,  elems 8-15 -> K=16..23
//  lanes 16-31: row M=lane-16,elems 0-7 -> K=8..15, elems 8-15 -> K=24..31
__device__ inline v16bf load_a_frag(const bhalf_t* base, int ld, int lane) {
  int row = lane & 15;
  int kh  = (lane >> 4) << 3;             // 0 or 8
  const bhalf_t* p = base + (size_t)row * ld + kh;
  v8bf lo = *(const v8bf*)(p);
  v8bf hi = *(const v8bf*)(p + 16);
  v16bf a;
#pragma unroll
  for (int i = 0; i < 8; ++i) { a[i] = lo[i]; a[i + 8] = hi[i]; }
  return a;
}

// B-matrix 32x16 bf16 fragment: lane -> column N=lane%16, elems 0-7 -> K=kh..kh+7, elems 8-15 -> K=16+kh..16+kh+7 (kh = 0 or 8).
__device__ inline v16bf load_b_frag(const bhalf_t* base, int ld, int lane) {
  int n  = lane & 15;
  int kh = (lane >> 4) << 3;
  v16bf b;
#pragma unroll
  for (int i = 0; i < 8; ++i) {
    b[i]     = base[(size_t)(kh + i) * ld + n];
    b[i + 8] = base[(size_t)(16 + kh + i) * ld + n];
  }
  return b;
}

// ---------------- CDNA5 async copy to LDS ----------------
// Each lane copies 16B per instruction; 8 instructions = 4KB chunk.
__device__ inline void async_copy_4k(uint32_t lds_off, const char* g, int lane) {
  uint32_t l = lds_off + (uint32_t)lane * 16u;
  const char* p = g + (size_t)lane * 16;
#pragma unroll
  for (int i = 0; i < 8; ++i) {
    asm volatile("global_load_async_to_lds_b128 %0, %1, off"
                 :: "v"(l), "v"(p) : "memory");
    l += 512u;
    p += 512;
  }
}
__device__ inline void async_wait_le8() { asm volatile("s_wait_asynccnt 0x8" ::: "memory"); }
__device__ inline void async_wait_0()   { asm volatile("s_wait_asynccnt 0x0" ::: "memory"); }
__device__ inline void ds_wait_0()      { asm volatile("s_wait_dscnt 0x0"    ::: "memory"); }

// ---------------- LayerNorm over channels (wave per position) ----------------

template <int C>
__global__ __launch_bounds__(256) void ln_kernel(
    const float* __restrict__ x, const float* __restrict__ g,
    const float* __restrict__ be, bhalf_t* __restrict__ outp) {
  int wave = blockIdx.x * (blockDim.x >> 5) + (threadIdx.x >> 5);
  int lane = threadIdx.x & 31;
  int b = wave / S, pos = wave % S;
  const float* xb = x + (size_t)b * C * S + pos;
  constexpr int PER = C / 32;
  float vals[PER];
  float s = 0.f;
#pragma unroll
  for (int i = 0; i < PER; ++i) {
    vals[i] = xb[(size_t)(lane + i * 32) * S];
    s += vals[i];
  }
#pragma unroll
  for (int off = 16; off; off >>= 1) s += __shfl_xor(s, off, 32);
  float mu = s * (1.0f / C);
  float v = 0.f;
#pragma unroll
  for (int i = 0; i < PER; ++i) { float d = vals[i] - mu; v += d * d; }
#pragma unroll
  for (int off = 16; off; off >>= 1) v += __shfl_xor(v, off, 32);
  float rstd = rsqrtf(v * (1.0f / C) + 1e-6f);
  bhalf_t* o = outp + ((size_t)b * S + pos) * C;
#pragma unroll
  for (int i = 0; i < PER; ++i) {
    int c = lane + i * 32;
    o[c] = (bhalf_t)((vals[i] - mu) * rstd * g[c] + be[c]);
  }
}

// ---------------- Weight transpose f32 -> bf16 ----------------

__global__ void wtrans_kernel(const float* __restrict__ w,
                              bhalf_t* __restrict__ wt, int OUT, int IN) {
  int i = blockIdx.x * blockDim.x + threadIdx.x;
  if (i >= OUT * IN) return;
  int o = i / IN, c = i % IN;
  wt[(size_t)c * OUT + o] = (bhalf_t)w[i];
}

// ---------------- QKV projection GEMM (32x64 tile per wave) ----------------
// out tile (pos x o) = A(b,S,Kd) x Bt(Kd,INNER). mode 0: store (bh,pos,hd); 1: (bh,hd,pos)
// Block = 8 waves covering a 128(M) x 128(N) tile: 4 M-subwaves x 2 N-subwaves.

__global__ __launch_bounds__(256) void qkv_gemm_kernel(
    const bhalf_t* __restrict__ A, const bhalf_t* __restrict__ Bt,
    const float* __restrict__ bias, bhalf_t* __restrict__ out,
    int Kd, int mode, int nblk_m) {
  int lane = threadIdx.x & 31;
  int wid  = threadIdx.x >> 5;
  int b    = blockIdx.y;
  int bm = blockIdx.x % nblk_m;
  int bn = blockIdx.x / nblk_m;
  int m0 = bm * 128 + (wid & 3) * 32;
  int n0 = bn * 128 + (wid >> 2) * 64;
  const bhalf_t* Arow = A + ((size_t)b * S + m0) * Kd;
  v8f acc[2][4] = {};
  for (int k0 = 0; k0 < Kd; k0 += 32) {
    v16bf a0 = load_a_frag(Arow + k0, Kd, lane);
    v16bf a1 = load_a_frag(Arow + (size_t)16 * Kd + k0, Kd, lane);
    const bhalf_t* Bp = Bt + (size_t)k0 * INNER + n0;
    v16bf b0 = load_b_frag(Bp,      INNER, lane);
    v16bf b1 = load_b_frag(Bp + 16, INNER, lane);
    v16bf b2 = load_b_frag(Bp + 32, INNER, lane);
    v16bf b3 = load_b_frag(Bp + 48, INNER, lane);
    acc[0][0] = wmma_bf16(a0, b0, acc[0][0]);
    acc[0][1] = wmma_bf16(a0, b1, acc[0][1]);
    acc[0][2] = wmma_bf16(a0, b2, acc[0][2]);
    acc[0][3] = wmma_bf16(a0, b3, acc[0][3]);
    acc[1][0] = wmma_bf16(a1, b0, acc[1][0]);
    acc[1][1] = wmma_bf16(a1, b1, acc[1][1]);
    acc[1][2] = wmma_bf16(a1, b2, acc[1][2]);
    acc[1][3] = wmma_bf16(a1, b3, acc[1][3]);
  }
  int n = lane & 15;
  int rowhi = (lane >> 4) * 8;
#pragma unroll
  for (int ni = 0; ni < 4; ++ni) {
    int o = n0 + ni * 16 + n;
    int head = o >> 6, hd = o & 63;
    float bb = bias[o];
#pragma unroll
    for (int mi = 0; mi < 2; ++mi) {
#pragma unroll
      for (int j = 0; j < 8; ++j) {
        int pos = m0 + mi * 16 + rowhi + j;
        float val = acc[mi][ni][j] + bb;
        size_t dst;
        if (mode == 0) dst = ((size_t)(b * HEADS + head) * S + pos) * HD + hd;
        else           dst = ((size_t)(b * HEADS + head) * HD + hd) * S + pos;
        out[dst] = (bhalf_t)val;
      }
    }
  }
}

// ---------------- Flash attention with relative bias ----------------
// grid (QT/4, B*HEADS), block 128: one wave per 16-query tile.
// V chunks (32 keys x 64 hd = 4KB) double-buffered in LDS via async copies.

constexpr int WAVE_LDS = 2 * 4096 + 1024;   // 2 V buffers + P transpose buffer

__global__ __launch_bounds__(128) void attn_kernel(
    const bhalf_t* __restrict__ Q,   // (bh, pos, hd)
    const bhalf_t* __restrict__ Kt,  // (bh, hd, pos)
    const bhalf_t* __restrict__ V,   // (bh, pos, hd)
    const float* __restrict__ rel,   // (127*127, HEADS)
    bhalf_t* __restrict__ O) {       // (b, pos, INNER)
  __shared__ __align__(128) char smem[4 * WAVE_LDS];
  int lane = threadIdx.x & 31;
  int wid  = threadIdx.x >> 5;
  int qt   = blockIdx.x * 4 + wid;          // 0..143
  int bh   = blockIdx.y;                    // 0..31
  int b = bh >> 3, head = bh & 7;
  int qbase = qt * 16;

  char* mybase = smem + wid * WAVE_LDS;
  // Flat shared addresses: low 32 bits == LDS byte offset (ISA 10.2 aperture rule)
  uint32_t my_lds = (uint32_t)(uintptr_t)mybase;
  const bhalf_t* vptr[2] = { (const bhalf_t*)mybase, (const bhalf_t*)(mybase + 4096) };
  uint32_t vlds[2] = { my_lds, my_lds + 4096u };
  bhalf_t* pl = (bhalf_t*)(mybase + 8192);

  const bhalf_t* Qb = Q + ((size_t)bh * S + qbase) * HD;
  v16bf aq0 = load_a_frag(Qb,      HD, lane);   // hd 0..31
  v16bf aq1 = load_a_frag(Qb + 32, HD, lane);   // hd 32..63

  v8f o0 = {}, o1 = {}, o2 = {}, o3 = {};
  float mrow[8], lrow[8];
  int qy[8], qx[8];
  int rowhi = (lane >> 4) * 8;
#pragma unroll
  for (int j = 0; j < 8; ++j) {
    mrow[j] = -1e30f; lrow[j] = 0.f;
    int q = qbase + rowhi + j;
    qy[j] = q / Ww; qx[j] = q % Ww;
  }

  const float scale = 0.125f;               // 1/sqrt(64)
  const bhalf_t* Kb  = Kt + (size_t)bh * HD * S;
  const bhalf_t* Vbh = V + (size_t)bh * S * HD;

  // prologue: start V chunk 0 into buffer 0
  async_copy_4k(vlds[0], (const char*)Vbh, lane);
  int buf = 0;

  for (int kc = 0; kc < S; kc += 32) {
    bool more = (kc + 32) < S;
    if (more) {
      // start next V chunk into the other buffer
      async_copy_4k(vlds[buf ^ 1], (const char*)(Vbh + (size_t)(kc + 32) * HD), lane);
      // prefetch next K chunk (64 rows x 64B): one line per lane, two rows deep
      const bhalf_t* nk = Kb + kc + 32;
      __builtin_prefetch(nk + (size_t)lane * S, 0, 0);
      __builtin_prefetch(nk + (size_t)(lane + 32) * S, 0, 0);
    }

    // --- scores: two 16x16 tiles over 32 keys ---
    v8f s0 = {}, s1 = {};
    {
      v16bf bk0 = load_b_frag(Kb + kc,                   S, lane);
      v16bf bk1 = load_b_frag(Kb + (size_t)32 * S + kc,  S, lane);
      s0 = wmma_bf16(aq0, bk0, s0);
      s0 = wmma_bf16(aq1, bk1, s0);
      v16bf bk2 = load_b_frag(Kb + kc + 16,                  S, lane);
      v16bf bk3 = load_b_frag(Kb + (size_t)32 * S + kc + 16, S, lane);
      s1 = wmma_bf16(aq0, bk2, s1);
      s1 = wmma_bf16(aq1, bk3, s1);
    }
    // --- scale + relative bias + online softmax ---
    int k0p = kc + (lane & 15);
    int k1p = k0p + 16;
    int ky0 = k0p / Ww, kx0 = k0p % Ww;
    int ky1 = k1p / Ww, kx1 = k1p % Ww;
    float mnew[8];
#pragma unroll
    for (int j = 0; j < 8; ++j) {
      int i0 = ((qy[j] - ky0 + 63) * 127 + (qx[j] - kx0 + 63)) * HEADS + head;
      int i1 = ((qy[j] - ky1 + 63) * 127 + (qx[j] - kx1 + 63)) * HEADS + head;
      s0[j] = s0[j] * scale + rel[i0];
      s1[j] = s1[j] * scale + rel[i1];
      float mt = fmaxf(s0[j], s1[j]);
#pragma unroll
      for (int off = 8; off; off >>= 1) mt = fmaxf(mt, __shfl_xor(mt, off, 16));
      mnew[j] = fmaxf(mrow[j], mt);
    }
#pragma unroll
    for (int j = 0; j < 8; ++j) {
      float alpha = __expf(mrow[j] - mnew[j]);
      lrow[j] *= alpha;
      o0[j] *= alpha; o1[j] *= alpha; o2[j] *= alpha; o3[j] *= alpha;
      float p0 = __expf(s0[j] - mnew[j]);
      float p1 = __expf(s1[j] - mnew[j]);
      mrow[j] = mnew[j];
      float rs = p0 + p1;
#pragma unroll
      for (int off = 8; off; off >>= 1) rs += __shfl_xor(rs, off, 16);
      lrow[j] += rs;
      // P -> LDS row-major 16x32 bf16 (wave-private region)
      pl[(rowhi + j) * 32 + (lane & 15)]      = (bhalf_t)p0;
      pl[(rowhi + j) * 32 + 16 + (lane & 15)] = (bhalf_t)p1;
    }
    ds_wait_0();
    v16bf pa = load_a_frag(pl, 32, lane);

    // current V chunk done (in-order completion); next chunk may stay in flight
    if (more) async_wait_le8(); else async_wait_0();
    const bhalf_t* vb = vptr[buf];
    o0 = wmma_bf16(pa, load_b_frag(vb +  0, HD, lane), o0);
    o1 = wmma_bf16(pa, load_b_frag(vb + 16, HD, lane), o1);
    o2 = wmma_bf16(pa, load_b_frag(vb + 32, HD, lane), o2);
    o3 = wmma_bf16(pa, load_b_frag(vb + 48, HD, lane), o3);
    buf ^= 1;
  }

  // epilogue: normalize + store (b, pos, head*64 + hd)
  bhalf_t* Ob = O + (size_t)b * S * INNER + (size_t)head * HD;
  int n = lane & 15;
#pragma unroll
  for (int j = 0; j < 8; ++j) {
    float inv = 1.0f / lrow[j];
    int pos = qbase + rowhi + j;
    bhalf_t* dst = Ob + (size_t)pos * INNER;
    dst[n]      = (bhalf_t)(o0[j] * inv);
    dst[16 + n] = (bhalf_t)(o1[j] * inv);
    dst[32 + n] = (bhalf_t)(o2[j] * inv);
    dst[48 + n] = (bhalf_t)(o3[j] * inv);
  }
}

// ---------------- Output projection + residual (32x64 tile per wave) ----------------

__global__ __launch_bounds__(256) void proj_kernel(
    const bhalf_t* __restrict__ Ob,   // (B,S,INNER)
    const bhalf_t* __restrict__ WpT,  // (INNER,CH)
    const float* __restrict__ bp, const float* __restrict__ x,
    float* __restrict__ out) {
  int lane = threadIdx.x & 31;
  int wid  = threadIdx.x >> 5;
  int b = blockIdx.y;
  int bm = blockIdx.x % 18;
  int bn = blockIdx.x / 18;
  int m0 = bm * 128 + (wid & 3) * 32;
  int n0 = bn * 128 + (wid >> 2) * 64;
  const bhalf_t* Arow = Ob + ((size_t)b * S + m0) * INNER;
  v8f acc[2][4] = {};
  for (int k0 = 0; k0 < INNER; k0 += 32) {
    v16bf a0 = load_a_frag(Arow + k0, INNER, lane);
    v16bf a1 = load_a_frag(Arow + (size_t)16 * INNER + k0, INNER, lane);
    const bhalf_t* Bp = WpT + (size_t)k0 * CH + n0;
    v16bf b0 = load_b_frag(Bp,      CH, lane);
    v16bf b1 = load_b_frag(Bp + 16, CH, lane);
    v16bf b2 = load_b_frag(Bp + 32, CH, lane);
    v16bf b3 = load_b_frag(Bp + 48, CH, lane);
    acc[0][0] = wmma_bf16(a0, b0, acc[0][0]);
    acc[0][1] = wmma_bf16(a0, b1, acc[0][1]);
    acc[0][2] = wmma_bf16(a0, b2, acc[0][2]);
    acc[0][3] = wmma_bf16(a0, b3, acc[0][3]);
    acc[1][0] = wmma_bf16(a1, b0, acc[1][0]);
    acc[1][1] = wmma_bf16(a1, b1, acc[1][1]);
    acc[1][2] = wmma_bf16(a1, b2, acc[1][2]);
    acc[1][3] = wmma_bf16(a1, b3, acc[1][3]);
  }
  int n = lane & 15, rowhi = (lane >> 4) * 8;
#pragma unroll
  for (int ni = 0; ni < 4; ++ni) {
    int co = n0 + ni * 16 + n;
    float bb = bp[co];
#pragma unroll
    for (int mi = 0; mi < 2; ++mi) {
#pragma unroll
      for (int j = 0; j < 8; ++j) {
        int pos = m0 + mi * 16 + rowhi + j;
        size_t idx = ((size_t)b * CH + co) * S + pos;
        out[idx] = x[idx] + acc[mi][ni][j] + bb;
      }
    }
  }
}

// ---------------- Host launcher ----------------

extern "C" void kernel_launch(void* const* d_in, const int* in_sizes, int n_in,
                              void* d_out, int out_size, void* d_ws, size_t ws_size,
                              hipStream_t stream) {
  (void)in_sizes; (void)n_in; (void)out_size; (void)ws_size;
  const float* x   = (const float*)d_in[0];
  const float* ctx = (const float*)d_in[1];
  const float* gx  = (const float*)d_in[2];
  const float* bx  = (const float*)d_in[3];
  const float* gc  = (const float*)d_in[4];
  const float* bc  = (const float*)d_in[5];
  const float* wq  = (const float*)d_in[6];
  const float* bq  = (const float*)d_in[7];
  const float* wk  = (const float*)d_in[8];
  const float* bk  = (const float*)d_in[9];
  const float* wv  = (const float*)d_in[10];
  const float* bv  = (const float*)d_in[11];
  const float* wp  = (const float*)d_in[12];
  const float* bp  = (const float*)d_in[13];
  const float* rel = (const float*)d_in[14];
  float* out = (float*)d_out;

  char* ws = (char*)d_ws;
  size_t off = 0;
  auto take = [&](size_t elems) -> bhalf_t* {
    bhalf_t* p = (bhalf_t*)(ws + off);
    off += (elems * sizeof(bhalf_t) + 255) & ~(size_t)255;
    return p;
  };
  bhalf_t* xn  = take((size_t)Bn * S * CH);
  bhalf_t* cn  = take((size_t)Bn * S * CTX);
  bhalf_t* wqT = take((size_t)CH * INNER);
  bhalf_t* wkT = take((size_t)CTX * INNER);
  bhalf_t* wvT = take((size_t)CTX * INNER);
  bhalf_t* wpT = take((size_t)INNER * CH);
  bhalf_t* Qb  = take((size_t)Bn * HEADS * S * HD);
  bhalf_t* Kb  = take((size_t)Bn * HEADS * S * HD);
  bhalf_t* Vb  = take((size_t)Bn * HEADS * S * HD);
  bhalf_t* Ob  = take((size_t)Bn * S * INNER);

  // 1) LayerNorms
  ln_kernel<CH ><<<dim3(Bn * S / 8), 256, 0, stream>>>(x,   gx, bx, xn);
  ln_kernel<CTX><<<dim3(Bn * S / 8), 256, 0, stream>>>(ctx, gc, bc, cn);

  // 2) Weight transpose + bf16 convert
  wtrans_kernel<<<(INNER * CH  + 255) / 256, 256, 0, stream>>>(wq, wqT, INNER, CH);
  wtrans_kernel<<<(INNER * CTX + 255) / 256, 256, 0, stream>>>(wk, wkT, INNER, CTX);
  wtrans_kernel<<<(INNER * CTX + 255) / 256, 256, 0, stream>>>(wv, wvT, INNER, CTX);
  wtrans_kernel<<<(CH * INNER  + 255) / 256, 256, 0, stream>>>(wp, wpT, CH, INNER);

  // 3) Q,K,V projections: 128x128 block tile, 32x64 per wave
  dim3 ggrid((S / 128) * (INNER / 128), Bn);   // (72, 4)
  qkv_gemm_kernel<<<ggrid, 256, 0, stream>>>(xn, wqT, bq, Qb, CH,  0, S / 128);
  qkv_gemm_kernel<<<ggrid, 256, 0, stream>>>(cn, wkT, bk, Kb, CTX, 1, S / 128);
  qkv_gemm_kernel<<<ggrid, 256, 0, stream>>>(cn, wvT, bv, Vb, CTX, 0, S / 128);

  // 4) Flash attention with relative bias (async V double-buffering in LDS)
  attn_kernel<<<dim3(QT / 4, Bn * HEADS), 128, 0, stream>>>(Qb, Kb, Vb, rel, Ob);

  // 5) Output projection + residual
  proj_kernel<<<dim3((S / 128) * (CH / 128), Bn), 256, 0, stream>>>(Ob, wpT, bp, x, out);
}
